// RPN_52810917872358
// MI455X (gfx1250) — matrix operand order repaired and checked by reading, not compiled, not run on hardware
//
#include <hip/hip_runtime.h>
#include <hip/hip_bf16.h>

// ---------------------------------------------------------------------------
// RPN proposal pipeline for MI455X (gfx1250, wave32, WMMA).
// Conv1 (3x3, 512->512) is 90.6 GFLOP -> implicit-im2col GEMM on
// v_wmma_f32_16x16x32_f16 with double-buffered LDS and
// global_load_async_to_lds_b128 staging.  Everything else is BW-trivial.
// ---------------------------------------------------------------------------

typedef _Float16 v16h __attribute__((ext_vector_type(16)));
typedef _Float16 v8h  __attribute__((ext_vector_type(8)));
typedef float    v8f  __attribute__((ext_vector_type(8)));
typedef int      v4i  __attribute__((vector_size(16)));   // matches builtin proto

#define AS1 __attribute__((address_space(1)))
#define AS3 __attribute__((address_space(3)))

#define HH      120
#define WWID    160
#define CIN     512
#define COUT    512
#define M_TOT   (HH * WWID)        // 19200 = 150 * 128
#define K_TOT   (CIN * 9)          // 4608
#define PH      (HH + 2)
#define PW      (WWID + 2)
#define NANCH   9
#define N_ALL   (M_TOT * NANCH)    // 172800
#define PRE_NMS 6000
#define POST_NMS 300
#define CAPK    8192
#define NBINS   4096
#define NEGF    (-1000000000.0f)
#define NEGH2   (-500000000.0f)

#if __has_builtin(__builtin_amdgcn_global_load_async_to_lds_b128)
#define ASYNC_LDS 1
#else
#define ASYNC_LDS 0
#endif

#if ASYNC_LDS
// 16B global -> LDS async copy (ASYNCcnt-tracked, no VGPR round trip)
__device__ __forceinline__ void async_cp16(const _Float16* g, _Float16* l) {
  __builtin_amdgcn_global_load_async_to_lds_b128(
      (AS1 v4i*)g, (AS3 v4i*)l, 0, 0);
}
#endif

// Anchors from generate_anchors(16,(.5,1,2),(8,16,32)) — fixed constants.
__device__ __constant__ float ANC[9][4] = {
    { -84.f,  -40.f,  99.f,  55.f},
    {-176.f,  -88.f, 191.f, 103.f},
    {-360.f, -184.f, 375.f, 199.f},
    { -56.f,  -56.f,  71.f,  71.f},
    {-120.f, -120.f, 135.f, 135.f},
    {-248.f, -248.f, 263.f, 263.f},
    { -36.f,  -80.f,  51.f,  95.f},
    { -80.f, -168.f,  95.f, 183.f},
    {-168.f, -344.f, 183.f, 359.f}};

// ---------------------------------------------------------------------------
// 1) weights f32[o][i][ky][kx] -> f16 B[o][ (ky*3+kx)*512 + i ]
// ---------------------------------------------------------------------------
__global__ __launch_bounds__(256) void k_convert_weights(
    const float* __restrict__ w, _Float16* __restrict__ Bt) {
  int idx = blockIdx.x * 256 + threadIdx.x;
  if (idx >= COUT * K_TOT) return;
  int o   = idx / K_TOT;
  int r   = idx - o * K_TOT;
  int kyx = r >> 9;
  int ci  = r & 511;
  Bt[idx] = (_Float16)w[o * K_TOT + ci * 9 + kyx];
}

// ---------------------------------------------------------------------------
// 2) features f32[c][y][x] -> zero-padded f16 P[yy][xx][c]
// ---------------------------------------------------------------------------
__global__ __launch_bounds__(256) void k_pad_features(
    const float* __restrict__ f, _Float16* __restrict__ P) {
  int idx = blockIdx.x * 256 + threadIdx.x;
  if (idx >= PH * PW * CIN) return;
  int c   = idx & 511;
  int rem = idx >> 9;
  int xx  = rem % PW;
  int yy  = rem / PW;
  float v = 0.f;
  if (yy >= 1 && yy <= HH && xx >= 1 && xx <= WWID)
    v = f[c * M_TOT + (yy - 1) * WWID + (xx - 1)];
  P[idx] = (_Float16)v;
}

// ---------------------------------------------------------------------------
// 3) Conv1 as WMMA GEMM:  X[m][n] = relu( sum_k A[m][k] B[n][k] + bias[n] )
//    128x128 block tile, 8 waves of 32x64, K step 32, double-buffered LDS.
//    Staging: each thread moves 2x16B per matrix per K-step (coalesced),
//    via global_load_async_to_lds_b128 when the builtin exists.
// ---------------------------------------------------------------------------
__global__ __launch_bounds__(256) void k_gemm_conv(
    const _Float16* __restrict__ P, const _Float16* __restrict__ Bt,
    const float* __restrict__ bias, float* __restrict__ X) {
  __shared__ __align__(32) _Float16 sA[2][128 * 32];
  __shared__ __align__(32) _Float16 sB[2][128 * 32];

  const int m0   = blockIdx.x * 128;
  const int n0   = blockIdx.y * 128;
  const int tid  = threadIdx.x;
  const int lane = tid & 31;
  const int wave = tid >> 5;
  const int wm   = (wave & 3) * 32;   // wave M offset in tile
  const int wn   = (wave >> 2) * 64;  // wave N offset in tile

  // ---- staging geometry: thread -> (row, two 8-half parts) ----------------
  const int srow  = tid >> 1;          // 0..127
  const int part0 = (tid & 1) * 2;     // parts {0,1} or {2,3}, 8 halfs each
  const int sy = (m0 + srow) / WWID;
  const int sx = (m0 + srow) - sy * WWID;
  const _Float16* gA = P + (size_t)(sy * PW + sx) * CIN;      // + poff + part*8
  const _Float16* gB = Bt + (size_t)(n0 + srow) * K_TOT;      // + kk   + part*8
  const int ldsRow = srow * 32;

  v8f acc[2][4] = {};

  const int mrow = lane & 15;
  const int aSel = (lane < 16) ? 0 : 16;   // A K-half select (contiguous, matches B)
  const int bSel = (lane < 16) ? 0 : 16;   // B K-half select (contiguous, matches A)

  // stage tile for K-chunk kk into buffer buf
  auto stage = [&](int buf, int kk) {
    const int kyx   = kk >> 9;
    const int ky    = kyx / 3;
    const int kx    = kyx - ky * 3;
    const int poff  = (ky * PW + kx) * CIN + (kk - (kyx << 9));
#if ASYNC_LDS
#pragma unroll
    for (int q = 0; q < 2; ++q) {
      int p8 = (part0 + q) * 8;
      async_cp16(gA + poff + p8, &sA[buf][ldsRow + p8]);
      async_cp16(gB + kk + p8, &sB[buf][ldsRow + p8]);
    }
#else
    v8h ra[2], rb[2];
#pragma unroll
    for (int q = 0; q < 2; ++q) {
      int p8 = (part0 + q) * 8;
      ra[q] = *(const v8h*)(gA + poff + p8);
      rb[q] = *(const v8h*)(gB + kk + p8);
    }
#pragma unroll
    for (int q = 0; q < 2; ++q) {
      int p8 = (part0 + q) * 8;
      *(v8h*)(&sA[buf][ldsRow + p8]) = ra[q];
      *(v8h*)(&sB[buf][ldsRow + p8]) = rb[q];
    }
    if (kk + 32 < K_TOT) {
      __builtin_prefetch(gB + kk + 32, 0, 1);
      __builtin_prefetch(gA + poff + 32, 0, 1);
    }
#endif
  };

  stage(0, 0);
  int cur = 0;
  for (int kk = 0; kk < K_TOT; kk += 32) {
#if ASYNC_LDS
#if __has_builtin(__builtin_amdgcn_s_wait_asynccnt)
    __builtin_amdgcn_s_wait_asynccnt(0);
#else
    asm volatile("s_wait_asynccnt 0x0" ::: "memory");
#endif
#endif
    __syncthreads();                    // buffer `cur` ready for all waves
    if (kk + 32 < K_TOT) stage(cur ^ 1, kk + 32);  // overlap with compute

#pragma unroll
    for (int im = 0; im < 2; ++im) {
      const _Float16* arow = &sA[cur][(wm + im * 16 + mrow) * 32];
      v8h lo = *(const v8h*)(arow + aSel);        // K 0..7  / 16..23
      v8h hi = *(const v8h*)(arow + aSel + 8);    // K 8..15 / 24..31
      v16h afrag = __builtin_shufflevector(lo, hi, 0, 1, 2, 3, 4, 5, 6, 7,
                                           8, 9, 10, 11, 12, 13, 14, 15);
#pragma unroll
      for (int in = 0; in < 4; ++in) {
        const _Float16* brow = &sB[cur][(wn + in * 16 + mrow) * 32 + bSel];
        v16h bfrag = *(const v16h*)brow;
        acc[im][in] = __builtin_amdgcn_wmma_f32_16x16x32_f16(
            false, afrag, false, bfrag, (short)0, acc[im][in], false, false);
      }
    }
    cur ^= 1;
  }

  // epilogue: bias + relu, D layout: VGPR r -> M = r + (lane<16?0:8), N = lane&15
#pragma unroll
  for (int im = 0; im < 2; ++im)
#pragma unroll
    for (int in = 0; in < 4; ++in) {
      int n = n0 + wn + in * 16 + mrow;
      float b = bias[n];
#pragma unroll
      for (int r = 0; r < 8; ++r) {
        int m = m0 + wm + im * 16 + r + ((lane < 16) ? 0 : 8);
        float v = acc[im][in][r] + b;
        X[(size_t)m * COUT + n] = v > 0.f ? v : 0.f;
      }
    }
}

// ---------------------------------------------------------------------------
// 4) 1x1 heads + softmax(18) + anchor decode + clip + min-size filter
// ---------------------------------------------------------------------------
__global__ __launch_bounds__(256) void k_heads_decode(
    const float* __restrict__ X, const float* __restrict__ sw,
    const float* __restrict__ sb, const float* __restrict__ bw,
    const float* __restrict__ bb, const float* __restrict__ iminfo,
    float* __restrict__ scores, float* __restrict__ props) {
  int m = blockIdx.x * 256 + threadIdx.x;
  if (m >= M_TOT) return;

  float accs[18], accd[36];
#pragma unroll
  for (int o = 0; o < 18; ++o) accs[o] = sb[o];
#pragma unroll
  for (int o = 0; o < 36; ++o) accd[o] = bb[o];

  const float* xr = X + (size_t)m * COUT;
#pragma unroll 4
  for (int c = 0; c < COUT; ++c) {
    float xv = xr[c];
#pragma unroll
    for (int o = 0; o < 18; ++o) accs[o] = fmaf(xv, sw[o * COUT + c], accs[o]);
#pragma unroll
    for (int o = 0; o < 36; ++o) accd[o] = fmaf(xv, bw[o * COUT + c], accd[o]);
  }

  float mx = accs[0];
#pragma unroll
  for (int o = 1; o < 18; ++o) mx = fmaxf(mx, accs[o]);
  float e[18], sum = 0.f;
#pragma unroll
  for (int o = 0; o < 18; ++o) { e[o] = __expf(accs[o] - mx); sum += e[o]; }
  float inv = 1.f / sum;

  int y = m / WWID, x = m - y * WWID;
  float imh = iminfo[0], imw = iminfo[1], iscale = iminfo[2];
  float fx = (float)x * 16.f, fy = (float)y * 16.f;

#pragma unroll
  for (int a = 0; a < 9; ++a) {
    float x1 = ANC[a][0] + fx, y1 = ANC[a][1] + fy;
    float x2 = ANC[a][2] + fx, y2 = ANC[a][3] + fy;
    float w  = x2 - x1 + 1.f, h = y2 - y1 + 1.f;
    float cx = x1 + 0.5f * w, cy = y1 + 0.5f * h;
    float dx = accd[a * 4 + 0], dy = accd[a * 4 + 1];
    float dw = accd[a * 4 + 2], dh = accd[a * 4 + 3];
    float pcx = dx * w + cx, pcy = dy * h + cy;
    float pw = __expf(dw) * w, ph = __expf(dh) * h;
    float px1 = pcx - 0.5f * pw, py1 = pcy - 0.5f * ph;
    float px2 = pcx + 0.5f * pw, py2 = pcy + 0.5f * ph;
    px1 = fminf(fmaxf(px1, 0.f), imw - 1.f);
    px2 = fminf(fmaxf(px2, 0.f), imw - 1.f);
    py1 = fminf(fmaxf(py1, 0.f), imh - 1.f);
    py2 = fminf(fmaxf(py2, 0.f), imh - 1.f);
    float ws_ = px2 - px1 + 1.f, hs_ = py2 - py1 + 1.f;
    bool keep = (ws_ >= 16.f * iscale) && (hs_ >= 16.f * iscale);
    int n = m * 9 + a;
    scores[n] = keep ? e[9 + a] * inv : NEGF;
    props[n * 4 + 0] = px1; props[n * 4 + 1] = py1;
    props[n * 4 + 2] = px2; props[n * 4 + 3] = py2;
  }
}

// ------------------------- top-k machinery ---------------------------------
__device__ __forceinline__ unsigned orderKey(float f) {
  unsigned u = __float_as_uint(f);
  return (u & 0x80000000u) ? ~u : (u | 0x80000000u);
}
__device__ __forceinline__ float unorderKey(unsigned s) {
  unsigned u = (s & 0x80000000u) ? (s & 0x7FFFFFFFu) : ~s;
  return __uint_as_float(u);
}

__global__ __launch_bounds__(256) void k_topk_init(
    unsigned* __restrict__ hist, int* __restrict__ cnt,
    unsigned long long* __restrict__ keys) {
  int i = blockIdx.x * 256 + threadIdx.x;
  if (i < NBINS) hist[i] = 0u;
  if (i < 8) cnt[i] = 0;
  if (i < CAPK) keys[i] = 0ull;    // padding key, sorts to the bottom
}

__global__ __launch_bounds__(256) void k_hist(
    const float* __restrict__ scores, unsigned* __restrict__ hist) {
  int i = blockIdx.x * 256 + threadIdx.x;
  if (i >= N_ALL) return;
  atomicAdd(&hist[orderKey(scores[i]) >> 20], 1u);
}

__global__ void k_find_cut(const unsigned* __restrict__ hist,
                           int* __restrict__ cnt) {
  if (threadIdx.x == 0 && blockIdx.x == 0) {
    int cum = 0, cut = 0;
    for (int b = NBINS - 1; b >= 0; --b) {
      cum += (int)hist[b];
      if (cum >= PRE_NMS) { cut = b; break; }
    }
    cnt[0] = cut;
  }
}

__global__ __launch_bounds__(256) void k_compact(
    const float* __restrict__ scores, int* __restrict__ cnt,
    unsigned long long* __restrict__ keys) {
  int i = blockIdx.x * 256 + threadIdx.x;
  if (i >= N_ALL) return;
  unsigned ok = orderKey(scores[i]);
  if ((int)(ok >> 20) >= cnt[0]) {
    int pos = atomicAdd(&cnt[1], 1);
    if (pos < CAPK)   // (score, ~idx): descending sort => score desc, idx asc
      keys[pos] = ((unsigned long long)ok << 32) | (0xFFFFFFFFu - (unsigned)i);
  }
}

// single-WG bitonic sort of 8192 u64 keys, exactly 64 KB LDS
__global__ __launch_bounds__(1024) void k_sort(
    unsigned long long* __restrict__ keys) {
  __shared__ unsigned long long sk[CAPK];
  int tid = threadIdx.x;
  for (int i = tid; i < CAPK; i += 1024) sk[i] = keys[i];
  __syncthreads();
  for (unsigned k = 2; k <= CAPK; k <<= 1) {
    for (unsigned j = k >> 1; j > 0; j >>= 1) {
      for (unsigned t = tid; t < CAPK / 2; t += 1024) {
        unsigned i = ((t & ~(j - 1)) << 1) | (t & (j - 1));
        unsigned p = i | j;
        bool asc = (i & k) != 0;           // top stage: asc=false => descending
        unsigned long long a = sk[i], b = sk[p];
        if (asc ? (a > b) : (a < b)) { sk[i] = b; sk[p] = a; }
      }
      __syncthreads();
    }
  }
  for (int i = tid; i < CAPK; i += 1024) keys[i] = sk[i];
}

__global__ __launch_bounds__(256) void k_gather(
    const unsigned long long* __restrict__ keys,
    const float* __restrict__ props, float* __restrict__ sboxes,
    float* __restrict__ sarea, float* __restrict__ sscore) {
  int i = blockIdx.x * 256 + threadIdx.x;
  if (i >= PRE_NMS) return;
  unsigned long long key = keys[i];
  float sc; unsigned idx;
  if (key == 0ull) { sc = NEGF; idx = 0; }
  else {
    sc  = unorderKey((unsigned)(key >> 32));
    idx = 0xFFFFFFFFu - (unsigned)(key & 0xFFFFFFFF);
    if (idx >= (unsigned)N_ALL) idx = 0;
  }
  float b0 = props[idx * 4 + 0], b1 = props[idx * 4 + 1];
  float b2 = props[idx * 4 + 2], b3 = props[idx * 4 + 3];
  sboxes[i * 4 + 0] = b0; sboxes[i * 4 + 1] = b1;
  sboxes[i * 4 + 2] = b2; sboxes[i * 4 + 3] = b3;
  sarea[i]  = (b2 - b0 + 1.f) * (b3 - b1 + 1.f);
  sscore[i] = sc;
}

// ---------------------------------------------------------------------------
// 5) NMS over sorted list (equivalent to reference argmax-scan) + ROI write
// ---------------------------------------------------------------------------
__global__ __launch_bounds__(1024) void k_nms(
    const float* __restrict__ sboxes, const float* __restrict__ sarea,
    const float* __restrict__ sscore, float* __restrict__ out) {
  __shared__ unsigned char flag[PRE_NMS];
  __shared__ int s_cur;
  __shared__ int s_ptr;
  int tid = threadIdx.x;
  for (int j = tid; j < PRE_NMS; j += 1024) flag[j] = 0;
  if (tid == 0) s_ptr = 0;
  __syncthreads();

  for (int s = 0; s < POST_NMS; ++s) {
    if (tid == 0) {
      int p = s_ptr;
      while (p < PRE_NMS && (flag[p] || sscore[p] <= NEGH2)) ++p;
      s_ptr = p;
      s_cur = (p < PRE_NMS) ? p : -1;
    }
    __syncthreads();
    int cur = s_cur;
    if (cur >= 0) {
      float bx0 = sboxes[cur * 4 + 0], by0 = sboxes[cur * 4 + 1];
      float bx1 = sboxes[cur * 4 + 2], by1 = sboxes[cur * 4 + 3];
      float ar  = sarea[cur];
      for (int j = tid; j < PRE_NMS; j += 1024) {
        if (!flag[j]) {
          float xx1 = fmaxf(bx0, sboxes[j * 4 + 0]);
          float yy1 = fmaxf(by0, sboxes[j * 4 + 1]);
          float xx2 = fminf(bx1, sboxes[j * 4 + 2]);
          float yy2 = fminf(by1, sboxes[j * 4 + 3]);
          float inter = fmaxf(xx2 - xx1 + 1.f, 0.f) * fmaxf(yy2 - yy1 + 1.f, 0.f);
          float iou = inter / (ar + sarea[j] - inter);
          if (iou > 0.7f) flag[j] = 1;     // flags cur itself too -> ptr advances
        }
      }
      if (tid == 0) {
        float* o = out + s * 5;
        o[0] = 0.f;
        o[1] = bx0; o[2] = by0; o[3] = bx1; o[4] = by1;
      }
    } else if (tid == 0) {
      float* o = out + s * 5;
      o[0] = o[1] = o[2] = o[3] = o[4] = 0.f;
    }
    __syncthreads();
  }
}

// ---------------------------------------------------------------------------
// launcher
// ---------------------------------------------------------------------------
extern "C" void kernel_launch(void* const* d_in, const int* in_sizes, int n_in,
                              void* d_out, int out_size, void* d_ws, size_t ws_size,
                              hipStream_t stream) {
  const float* features = (const float*)d_in[0];
  const float* im_info  = (const float*)d_in[1];
  const float* conv1_w  = (const float*)d_in[2];
  const float* conv1_b  = (const float*)d_in[3];
  const float* score_w  = (const float*)d_in[4];
  const float* score_b  = (const float*)d_in[5];
  const float* bbox_w   = (const float*)d_in[6];
  const float* bbox_b   = (const float*)d_in[7];
  float* out = (float*)d_out;

  char* ws = (char*)d_ws;
  size_t off = 0;
  auto alloc = [&](size_t bytes) {
    size_t p = off;
    off = (off + bytes + 255) & ~(size_t)255;
    return p;
  };
  _Float16* Bt   = (_Float16*)(ws + alloc((size_t)COUT * K_TOT * 2));   // 4.7 MB
  _Float16* P    = (_Float16*)(ws + alloc((size_t)PH * PW * CIN * 2));  // 20.2 MB
  float*    X    = (float*)(ws + alloc((size_t)M_TOT * COUT * 4));      // 39.3 MB
  float*    scr  = (float*)(ws + alloc((size_t)N_ALL * 4));
  float*    prp  = (float*)(ws + alloc((size_t)N_ALL * 4 * 4));
  unsigned* hist = (unsigned*)(ws + alloc(NBINS * 4));
  int*      cnt  = (int*)(ws + alloc(64));
  unsigned long long* keys = (unsigned long long*)(ws + alloc(CAPK * 8));
  float*    sbx  = (float*)(ws + alloc(PRE_NMS * 4 * 4));
  float*    sar  = (float*)(ws + alloc(PRE_NMS * 4));
  float*    ssc  = (float*)(ws + alloc(PRE_NMS * 4));

  k_convert_weights<<<(COUT * K_TOT + 255) / 256, 256, 0, stream>>>(conv1_w, Bt);
  k_pad_features<<<(PH * PW * CIN + 255) / 256, 256, 0, stream>>>(features, P);
  k_gemm_conv<<<dim3(M_TOT / 128, COUT / 128), 256, 0, stream>>>(P, Bt, conv1_b, X);
  k_heads_decode<<<(M_TOT + 255) / 256, 256, 0, stream>>>(
      X, score_w, score_b, bbox_w, bbox_b, im_info, scr, prp);
  k_topk_init<<<(CAPK + 255) / 256, 256, 0, stream>>>(hist, cnt, keys);
  k_hist<<<(N_ALL + 255) / 256, 256, 0, stream>>>(scr, hist);
  k_find_cut<<<1, 32, 0, stream>>>(hist, cnt);
  k_compact<<<(N_ALL + 255) / 256, 256, 0, stream>>>(scr, cnt, keys);
  k_sort<<<1, 1024, 0, stream>>>(keys);
  k_gather<<<(PRE_NMS + 255) / 256, 256, 0, stream>>>(keys, prp, sbx, sar, ssc);
  k_nms<<<1, 1024, 0, stream>>>(sbx, sar, ssc, out);
}
